// ScoreNetGNN_30288109372085
// MI455X (gfx1250) — matrix operand order repaired and checked by reading, not compiled, not run on hardware
//
#include <hip/hip_runtime.h>
#include <hip/hip_bf16.h>

typedef __attribute__((ext_vector_type(16))) _Float16 v16h;
typedef __attribute__((ext_vector_type(8)))  _Float16 h8;
typedef __attribute__((ext_vector_type(8)))  float    v8f;

#define GPW    8               // graphs per workgroup (50000 % 8 == 0)
#define TNOBJ  7
#define NODES  (GPW * TNOBJ)   // 56
#define NR     64              // node rows padded to 4 WMMA tiles
#define EDGES  (GPW * 42)      // 336 = 21 exact 16-row tiles
#define HID    64
#define EMBW   32
#define LN_SIGMA 3.2188758248682006f   // ln(25)

// ---- WMMA operand builders (layouts per CDNA5 ISA 7.12.2) ------------------
// A 16x32 f16: lane m=lane%16; elements 0..7 <- k = 8*laneHi + ks*32 + {0..7}
//              elements 8..15 <- k = 16 + 8*laneHi + ks*32 + {0..7}
__device__ __forceinline__ v16h load_A(const _Float16* rowPtr, int laneHi, int ks) {
    h8 c0 = *(const h8*)(rowPtr + laneHi * 8 + ks * 32);
    h8 c1 = *(const h8*)(rowPtr + laneHi * 8 + 16 + ks * 32);
    v16h a;
    #pragma unroll
    for (int i = 0; i < 8; ++i) { a[i] = c0[i]; a[8 + i] = c1[i]; }
    return a;
}
// B 32x16 f16: lane n=lane%16; elements 0..7 <- k = 8*laneHi + ks*32 + {0..7}, 8..15 <- k = 16 + 8*laneHi + ks*32 + {0..7}
// (weights staged TRANSPOSED in LDS: colPtr = &Wt[col][0], contiguous in k)
__device__ __forceinline__ v16h load_B(const _Float16* colPtr, int laneHi, int ks) {
    h8 c0 = *(const h8*)(colPtr + laneHi * 8 + ks * 32);
    h8 c1 = *(const h8*)(colPtr + laneHi * 8 + 16 + ks * 32);
    v16h b;
    #pragma unroll
    for (int i = 0; i < 8; ++i) { b[i] = c0[i]; b[8 + i] = c1[i]; }
    return b;
}

// ---------------------------------------------------------------------------
// Prologue 1: Gaussian-Fourier time embedding
// ---------------------------------------------------------------------------
__global__ __launch_bounds__(256)
void time_embed_kernel(const float* __restrict__ t, const float* __restrict__ Wf,
                       const float* __restrict__ wt, const float* __restrict__ bt,
                       float* __restrict__ xs, int B) {
    int b = blockIdx.x * 256 + threadIdx.x;
    if (b >= B) return;
    float tv = t[b];
    float e[EMBW];
    #pragma unroll
    for (int k = 0; k < 16; ++k) {
        float p = tv * Wf[k] * 6.28318530717958647f;
        float s = __sinf(p), c = __cosf(p);
        e[k]      = s > 0.f ? s : 0.f;
        e[16 + k] = c > 0.f ? c : 0.f;
    }
    #pragma unroll 4
    for (int j = 0; j < EMBW; ++j) {
        float acc = bt[j];
        #pragma unroll
        for (int k = 0; k < EMBW; ++k) acc += e[k] * wt[k * EMBW + j];
        xs[(size_t)b * EMBW + j] = acc > 0.f ? acc : 0.f;
    }
}

// ---------------------------------------------------------------------------
// Prologue 2: init_feature = relu(omega@wi1+bi1)@wi2+bi2   (N x 64, f16 out)
// ---------------------------------------------------------------------------
__global__ __launch_bounds__(256)
void init_feat_kernel(const float* __restrict__ omega,
                      const float* __restrict__ wi1, const float* __restrict__ bi1,
                      const float* __restrict__ wi2, const float* __restrict__ bi2,
                      _Float16* __restrict__ feat, int N) {
    int i = blockIdx.x * 256 + threadIdx.x;
    if (i >= N) return;
    float o0 = omega[(size_t)i * 3 + 0];
    float o1 = omega[(size_t)i * 3 + 1];
    float o2 = omega[(size_t)i * 3 + 2];
    float h[HID];
    #pragma unroll 8
    for (int j = 0; j < HID; ++j) {
        float a = o0 * wi1[j] + o1 * wi1[HID + j] + o2 * wi1[2 * HID + j] + bi1[j];
        h[j] = a > 0.f ? a : 0.f;
    }
    #pragma unroll
    for (int j8 = 0; j8 < HID / 8; ++j8) {
        h8 o;
        #pragma unroll
        for (int q = 0; q < 8; ++q) {
            int j = j8 * 8 + q;
            float acc = bi2[j];
            #pragma unroll 8
            for (int k = 0; k < HID; ++k) acc += h[k] * wi2[k * HID + j];
            o[q] = (_Float16)acc;
        }
        *(h8*)&feat[(size_t)i * HID + j8 * 8] = o;   // global_store_b128
    }
}

// ---------------------------------------------------------------------------
// Fused EdgeConv layer (see round-0 analysis for the algebraic split).
// ---------------------------------------------------------------------------
template <int F_IN, int F_OUT, bool FINAL>
__global__ __launch_bounds__(256)
void edgeconv_kernel(const _Float16* __restrict__ featIn,   // N x F_IN
                     _Float16* __restrict__ featOut,        // N x 96 (non-final)
                     float* __restrict__ outFinal,          // N x 3 (final)
                     const float* __restrict__ wa,          // (2*F_IN) x 64
                     const float* __restrict__ ba,          // 64
                     const float* __restrict__ wb,          // 64 x (FINAL?3:64)
                     const float* __restrict__ bb,
                     const float* __restrict__ xs,          // B x 32
                     const float* __restrict__ t,           // B
                     int B) {
    constexpr int FOUT_REAL = FINAL ? 3 : F_OUT;
    constexpr int KSTEPS    = F_IN / 32;
    constexpr int CT        = F_OUT / 16;

    __shared__ __align__(16) _Float16 sF  [NR][F_IN];    // node features
    __shared__ __align__(16) _Float16 sWdT[HID][F_IN];   // (Wa_top-Wa_bot)^T
    __shared__ __align__(16) _Float16 sWsT[HID][F_IN];   // Wa_bot^T
    __shared__ __align__(16) _Float16 sWbT[F_OUT][HID];  // Wb^T (padded cols)
    __shared__ float sBa[HID];
    __shared__ float sBb[F_OUT];
    __shared__ __align__(16) _Float16 sU[NR][HID];
    __shared__ __align__(16) _Float16 sV[NR][HID];
    __shared__ __align__(16) _Float16 sM[EDGES][HID];    // relu(u_d+v_s), dst-grouped
    __shared__ __align__(16) _Float16 sMOh[FINAL ? 8 : EDGES * HID];
    __shared__ __align__(16) float    sMOf[FINAL ? EDGES * F_OUT : 8];
    __shared__ int sDstL[42], sSrcL[42];                 // per-graph edge LUT

    const int tid        = threadIdx.x;
    const int graphBase  = blockIdx.x * GPW;
    const int nodeBase   = graphBase * TNOBJ;
    const int graphsHere = min(GPW, B - graphBase);
    const int nodesHere  = graphsHere * TNOBJ;

    // ---- Phase 0: stage weights (transposed, f16) + features into LDS ----
    if (tid < 42) {
        int d = tid / 6, j = tid % 6;
        sDstL[tid] = d;
        sSrcL[tid] = j + (j >= d ? 1 : 0);
    }
    for (int i = tid; i < F_IN * HID; i += 256) {
        int k = i >> 6, j = i & 63;                      // HID == 64
        float top = wa[(size_t)k * HID + j];
        float bot = wa[(size_t)(F_IN + k) * HID + j];
        sWdT[j][k] = (_Float16)(top - bot);
        sWsT[j][k] = (_Float16)bot;
    }
    for (int i = tid; i < HID * F_OUT; i += 256) {
        int k = i / F_OUT, j = i % F_OUT;                // F_OUT pow2 -> shifts
        sWbT[j][k] = (_Float16)((j < FOUT_REAL) ? wb[(size_t)k * FOUT_REAL + j] : 0.f);
    }
    if (tid < HID)   sBa[tid] = ba[tid];
    if (tid < F_OUT) sBb[tid] = (tid < FOUT_REAL) ? bb[tid] : 0.f;
    for (int i = tid; i < NR * (F_IN / 8); i += 256) {   // 16B vector copies
        int r = i / (F_IN / 8), c8 = (i % (F_IN / 8)) * 8;
        h8 v;
        if (r < nodesHere) {
            v = *(const h8*)&featIn[(size_t)(nodeBase + r) * F_IN + c8];
        } else {
            v = (h8)((_Float16)0.f);                     // splat zero
        }
        *(h8*)&sF[r][c8] = v;
    }
    __syncthreads();

    const int lane   = tid & 31;
    const int wave   = tid >> 5;
    const int laneM  = lane & 15;
    const int laneHi = lane >> 4;

    // ---- Phase 1: U = X@Wd + ba ; V = X@Ws  (shared A, two accumulators) ----
    for (int tile = wave; tile < 16; tile += 8) {        // 4rt x 4ct, 2 per wave
        const int ct = tile & 3;
        const int rt = tile >> 2;
        v8f accU = {}, accV = {};
        #pragma unroll
        for (int ks = 0; ks < KSTEPS; ++ks) {
            v16h a  = load_A(&sF  [rt * 16 + laneM][0], laneHi, ks);
            v16h bd = load_B(&sWdT[ct * 16 + laneM][0], laneHi, ks);
            v16h bs = load_B(&sWsT[ct * 16 + laneM][0], laneHi, ks);
            accU = __builtin_amdgcn_wmma_f32_16x16x32_f16(false, a, false, bd,
                                                          (short)0, accU, false, false);
            accV = __builtin_amdgcn_wmma_f32_16x16x32_f16(false, a, false, bs,
                                                          (short)0, accV, false, false);
        }
        const int n = ct * 16 + laneM;
        const float biasU = sBa[n];
        #pragma unroll
        for (int r = 0; r < 8; ++r) {                    // C/D: VGPR r -> M=r(+8)
            int m = rt * 16 + r + laneHi * 8;
            sU[m][n] = (_Float16)(accU[r] + biasU);
            sV[m][n] = (_Float16)accV[r];
        }
    }
    __syncthreads();

    // ---- Phase 2: messages relu(u_d + v_s), dst-grouped rows (6 per node) ----
    for (int i = tid; i < EDGES * 8; i += 256) {         // h8 chunks
        int row = i >> 3, c8 = (i & 7) * 8;
        int g   = row / 42;
        int rem = row - g * 42;
        h8 u = *(const h8*)&sU[g * TNOBJ + sDstL[rem]][c8];
        h8 v = *(const h8*)&sV[g * TNOBJ + sSrcL[rem]][c8];
        h8 m;
        #pragma unroll
        for (int q = 0; q < 8; ++q) {
            _Float16 s = u[q] + v[q];
            m[q] = s > (_Float16)0.f ? s : (_Float16)0.f;
        }
        *(h8*)&sM[row][c8] = m;
    }
    __syncthreads();

    // ---- Phase 3: msg @ Wb + bb  (21 row-tiles x CT col-tiles) ----
    for (int tile = wave; tile < 21 * CT; tile += 8) {
        const int ct = tile % CT;
        const int rt = tile / CT;
        v8f acc = {};
        #pragma unroll
        for (int ks = 0; ks < 2; ++ks) {                 // K = 64
            v16h a = load_A(&sM  [rt * 16 + laneM][0], laneHi, ks);
            v16h b = load_B(&sWbT[ct * 16 + laneM][0], laneHi, ks);
            acc = __builtin_amdgcn_wmma_f32_16x16x32_f16(false, a, false, b,
                                                         (short)0, acc, false, false);
        }
        const int n    = ct * 16 + laneM;
        const float bv = sBb[n];
        #pragma unroll
        for (int r = 0; r < 8; ++r) {
            int m = rt * 16 + r + laneHi * 8;
            float vv = acc[r] + bv;
            if (FINAL) sMOf[m * F_OUT + n] = vv;
            else       sMOh[m * HID + n]   = (_Float16)vv;
        }
    }
    __syncthreads();

    // ---- Phase 4: 6-row max reduce + epilogue ----
    if (!FINAL) {
        for (int i = tid; i < NODES * 8; i += 256) {     // h8 chunks of 64 cols
            int n = i >> 3, c8 = (i & 7) * 8;
            if (n >= nodesHere) continue;
            const _Float16* base = &sMOh[(6 * n) * HID + c8];
            h8 m0 = *(const h8*)(base + 0 * HID);
            h8 m1 = *(const h8*)(base + 1 * HID);
            h8 m2 = *(const h8*)(base + 2 * HID);
            h8 m3 = *(const h8*)(base + 3 * HID);
            h8 m4 = *(const h8*)(base + 4 * HID);
            h8 m5 = *(const h8*)(base + 5 * HID);
            h8 o;
            #pragma unroll
            for (int q = 0; q < 8; ++q) {
                _Float16 mx = m0[q];
                mx = m1[q] > mx ? m1[q] : mx;
                mx = m2[q] > mx ? m2[q] : mx;
                mx = m3[q] > mx ? m3[q] : mx;
                mx = m4[q] > mx ? m4[q] : mx;
                mx = m5[q] > mx ? m5[q] : mx;
                o[q] = mx > (_Float16)0.f ? mx : (_Float16)0.f;   // relu
            }
            *(h8*)&featOut[(size_t)(nodeBase + n) * 96 + c8] = o;
        }
        // concat x_sigma: tile-whole-block semantics -> xs[globalNode % B]
        for (int i = tid; i < NODES * 4; i += 256) {     // 4 chunks of 8 cols
            int n = i >> 2, c8 = (i & 3) * 8;
            if (n >= nodesHere) continue;
            size_t gi = (size_t)(nodeBase + n);
            const float* src = xs + (size_t)((int)(gi % (size_t)B)) * EMBW + c8;
            h8 o;
            #pragma unroll
            for (int q = 0; q < 8; ++q) o[q] = (_Float16)src[q];
            *(h8*)&featOut[gi * 96 + HID + c8] = o;
        }
    } else {
        for (int i = tid; i < NODES * 3; i += 256) {
            int n = i / 3, c = i - n * 3;
            if (n >= nodesHere) continue;
            const float* base = &sMOf[(6 * n) * F_OUT + c];
            float mx = base[0];
            #pragma unroll
            for (int j = 1; j < 6; ++j) {
                float vv = base[j * F_OUT];
                mx = vv > mx ? vv : mx;
            }
            size_t gi  = (size_t)(nodeBase + n);
            float tv   = t[gi / TNOBJ];                  // repeat_interleave
            float stdv = __fsqrt_rn((__expf(2.f * tv * LN_SIGMA) - 1.f)
                                    / (2.f * LN_SIGMA));
            outFinal[gi * 3 + c] = mx / (stdv + 1e-7f);
        }
    }
}

// ---------------------------------------------------------------------------
extern "C" void kernel_launch(void* const* d_in, const int* in_sizes, int n_in,
                              void* d_out, int out_size, void* d_ws, size_t ws_size,
                              hipStream_t stream) {
    const float* omega = (const float*)d_in[0];
    const float* t     = (const float*)d_in[2];
    const float* Wf    = (const float*)d_in[4];
    const float* wi1   = (const float*)d_in[5];
    const float* bi1   = (const float*)d_in[6];
    const float* wi2   = (const float*)d_in[7];
    const float* bi2   = (const float*)d_in[8];
    const float* wt    = (const float*)d_in[9];
    const float* bt    = (const float*)d_in[10];
    const float* w1a   = (const float*)d_in[11];
    const float* b1a   = (const float*)d_in[12];
    const float* w1b   = (const float*)d_in[13];
    const float* b1b   = (const float*)d_in[14];
    const float* w2a   = (const float*)d_in[15];
    const float* b2a   = (const float*)d_in[16];
    const float* w2b   = (const float*)d_in[17];
    const float* b2b   = (const float*)d_in[18];
    const float* w3a   = (const float*)d_in[19];
    const float* b3a   = (const float*)d_in[20];
    const float* w3b   = (const float*)d_in[21];
    const float* b3b   = (const float*)d_in[22];

    const int B = in_sizes[2];           // 50000
    const int N = in_sizes[0] / 3;       // 350000

    char*  ws  = (char*)d_ws;
    size_t off = 0;
    auto carve = [&](size_t bytes) {
        void* p = ws + off;
        off = (off + bytes + 255) & ~(size_t)255;
        return p;
    };
    float*    xs    = (float*)   carve((size_t)B * EMBW * sizeof(float));
    _Float16* feat0 = (_Float16*)carve((size_t)N * HID * sizeof(_Float16));
    _Float16* featA = (_Float16*)carve((size_t)N * 96  * sizeof(_Float16));
    _Float16* featB = (_Float16*)carve((size_t)N * 96  * sizeof(_Float16));
    (void)ws_size; (void)n_in; (void)out_size;

    time_embed_kernel<<<(B + 255) / 256, 256, 0, stream>>>(t, Wf, wt, bt, xs, B);
    init_feat_kernel <<<(N + 255) / 256, 256, 0, stream>>>(omega, wi1, bi1, wi2, bi2,
                                                           feat0, N);

    const int nblk = (B + GPW - 1) / GPW;   // 6250
    edgeconv_kernel<64, 64, false><<<nblk, 256, 0, stream>>>(
        feat0, featA, nullptr, w1a, b1a, w1b, b1b, xs, t, B);
    edgeconv_kernel<96, 64, false><<<nblk, 256, 0, stream>>>(
        featA, featB, nullptr, w2a, b2a, w2b, b2b, xs, t, B);
    edgeconv_kernel<96, 16, true><<<nblk, 256, 0, stream>>>(
        featB, nullptr, (float*)d_out, w3a, b3a, w3b, b3b, xs, t, B);
}
